// CustomLSTM_8985071583762
// MI455X (gfx1250) — matrix operand order repaired and checked by reading, not compiled, not run on hardware
//
#include <hip/hip_runtime.h>

typedef __attribute__((ext_vector_type(16))) __bf16 v16bf;
typedef __attribute__((ext_vector_type(8)))  __bf16 v8bf;
typedef __attribute__((ext_vector_type(8)))  float  v8f;

namespace {
constexpr int B_ = 64;
constexpr int S_ = 1024;
constexpr int I_ = 256;
constexpr int H_ = 256;
constexpr int G_ = 4 * H_;               // 1024 gate columns
constexpr long BS_ = (long)B_ * S_;      // 65536 rows for the x-projection GEMM
}

__device__ __forceinline__ unsigned short f2bf(float f) {
  unsigned u = __float_as_uint(f);
  u += 0x7FFFu + ((u >> 16) & 1u);       // round-to-nearest-even
  return (unsigned short)(u >> 16);
}

__device__ __forceinline__ float sigmoidf_(float x) {
  return 1.0f / (1.0f + __expf(-x));
}

// ---- CDNA5 async copy (global -> LDS), tracked by ASYNCcnt ----
// GV addressing: VDST = 32-bit LDS byte offset, VADDR = 64-bit global address.
__device__ __forceinline__ void async_ld16(unsigned lds_byte_off, const void* g) {
  unsigned long long ga = (unsigned long long)g;
  asm volatile("global_load_async_to_lds_b128 %0, %1, off"
               :: "v"(lds_byte_off), "v"(ga) : "memory");
}
__device__ __forceinline__ void wait_async_zero() {
  asm volatile("s_wait_asynccnt 0x0" ::: "memory");
}

// A-fragment (16x32 bf16, ISA 7.12.2): lane L holds row m=(L&15);
// elements 0..7  = K[kk + 8*(L>>4) .. +7]
// elements 8..15 = K[kk + 16 + 8*(L>>4) .. +7]
__device__ __forceinline__ v16bf frag_a(const unsigned short* rowptr, int kk, int half) {
  v8bf lo = *(const v8bf*)(rowptr + kk + 8 * half);
  v8bf hi = *(const v8bf*)(rowptr + kk + 16 + 8 * half);
  return __builtin_shufflevector(lo, hi, 0, 1, 2, 3, 4, 5, 6, 7,
                                 8, 9, 10, 11, 12, 13, 14, 15);
}

// B-fragment (32x16 bf16). B = W^T, so column n of B is row n of W (contiguous K).
// lane L holds column n=(L&15); elements 0..7 = K[kk + 8*(L>>4) .. +7], elements 8..15 = K[kk + 16 + 8*(L>>4) .. +7].
__device__ __forceinline__ v16bf frag_b(const unsigned short* rowptr, int kk, int half) {
  v8bf lo = *(const v8bf*)(rowptr + kk + 8 * half);
  v8bf hi = *(const v8bf*)(rowptr + kk + 16 + 8 * half);
  return __builtin_shufflevector(lo, hi, 0, 1, 2, 3, 4, 5, 6, 7,
                                 8, 9, 10, 11, 12, 13, 14, 15);
}

// ---------------- fp32 -> bf16 conversion kernels ----------------

__global__ __launch_bounds__(256) void cvt_x_bf16(const float* __restrict__ src,
                                                  unsigned short* __restrict__ dst,
                                                  int n) {
  int i = (blockIdx.x * 256 + threadIdx.x) * 4;
  if (i + 3 < n) {
    float4 v = *(const float4*)(src + i);
    dst[i + 0] = f2bf(v.x);
    dst[i + 1] = f2bf(v.y);
    dst[i + 2] = f2bf(v.z);
    dst[i + 3] = f2bf(v.w);
  }
}

// Split W (G x (H+I), row-major) into bf16 W_h (G x H) and W_x (G x I).
__global__ __launch_bounds__(256) void cvt_w_bf16(const float* __restrict__ W,
                                                  unsigned short* __restrict__ Whb,
                                                  unsigned short* __restrict__ Wxb) {
  int idx = blockIdx.x * 256 + threadIdx.x;     // 0 .. G*(H+I)-1
  int g = idx >> 9;                             // row (H+I == 512)
  int c = idx & 511;                            // col
  unsigned short v = f2bf(W[idx]);
  if (c < H_) Whb[g * H_ + c] = v;
  else        Wxb[g * I_ + (c - H_)] = v;
}

// ---------------- Phase 1: x_proj = x @ W_x^T + b ----------------
// Block = 256 threads (8 waves). Block tile 64(M) x 64(N), full K=256 staged in LDS:
//   x tile  64x256 bf16 = 32 KB  (async copy, each global byte read once per block)
//   Wx tile 64x256 bf16 = 32 KB
// Wave (mq = w&1 -> 32 rows, nq = w>>1 -> 16 cols): 2 accumulators, B-frag reused 2x,
// entire K-loop runs out of LDS (ds_load_b128).
__global__ __launch_bounds__(256) void gemm_xproj(
    const unsigned short* __restrict__ xb,    // [BS][I] bf16
    const unsigned short* __restrict__ Wxb,   // [G][I]  bf16
    const float* __restrict__ bias,           // [G]
    float* __restrict__ xproj)                // [BS][G] f32
{
  __shared__ unsigned short xs[64 * 256];     // 32 KB
  __shared__ unsigned short ws[64 * 256];     // 32 KB

  const int tid  = threadIdx.x;
  const int lane = tid & 31;
  const int wave = tid >> 5;
  const int half = lane >> 4;
  const int l16  = lane & 15;
  const int mq = wave & 1;
  const int nq = wave >> 1;
  const int mrow0 = blockIdx.x * 64;
  const int n0    = blockIdx.y * 64;

  // LDS byte offsets (generic-pointer low 32 bits == wave-relative LDS address)
  const unsigned xs_base = (unsigned)(unsigned long long)(const void*)xs;
  const unsigned ws_base = (unsigned)(unsigned long long)(const void*)ws;

  // Stage both tiles: 2048 16-byte chunks each, 8 per thread per tile.
  for (int c = tid; c < 2048; c += 256) {
    const int row = c >> 5;
    const int cc  = (c & 31) * 8;
    async_ld16(xs_base + (unsigned)(row * 256 + cc) * 2,
               xb + (size_t)(mrow0 + row) * I_ + cc);
    async_ld16(ws_base + (unsigned)(row * 256 + cc) * 2,
               Wxb + (size_t)(n0 + row) * I_ + cc);
  }
  wait_async_zero();
  __syncthreads();

  v8f acc[2];
#pragma unroll
  for (int mt = 0; mt < 2; ++mt)
#pragma unroll
    for (int r = 0; r < 8; ++r) acc[mt][r] = 0.0f;

  const unsigned short* xrow = xs + (mq * 32 + l16) * 256;
  const unsigned short* wrow = ws + (nq * 16 + l16) * 256;

#pragma unroll
  for (int kk = 0; kk < I_; kk += 32) {
    v16bf bf = frag_b(wrow, kk, half);
    v16bf a0 = frag_a(xrow, kk, half);
    v16bf a1 = frag_a(xrow + 16 * 256, kk, half);
    acc[0] = __builtin_amdgcn_wmma_f32_16x16x32_bf16(
        false, a0, false, bf, (short)0, acc[0], false, false);
    acc[1] = __builtin_amdgcn_wmma_f32_16x16x32_bf16(
        false, a1, false, bf, (short)0, acc[1], false, false);
  }

  const int nglob = n0 + nq * 16 + l16;
  const float bv = bias[nglob];
#pragma unroll
  for (int mt = 0; mt < 2; ++mt) {
#pragma unroll
    for (int r = 0; r < 8; ++r) {
      const int m = mrow0 + mq * 32 + mt * 16 + r + 8 * half;
      xproj[(size_t)m * G_ + nglob] = acc[mt][r] + bv;
    }
  }
}

// ---------------- Phase 2: persistent single-workgroup LSTM scan ----------------
// 1024 threads = 32 waves on one WGP. h double-buffered in LDS (bf16), c in VGPRs.
// Tile groups: gid in 0..63 -> (mt = gid&3 : 16 batch rows, js = gid>>2 : 16 hidden cols).
// Wave w owns groups 2w, 2w+1; for each group it computes all 4 gate tiles, so the cell
// update happens entirely in registers (f,i,g,o land in the same lane/VGPR slots).
// x_proj is loaded into registers up front and folded in AFTER the K-loop, keeping the
// 32 scattered global loads in flight underneath the 32 WMMAs.
__global__ __launch_bounds__(1024) void lstm_scan(
    const unsigned short* __restrict__ Whb,   // [G][H] bf16
    const float* __restrict__ xproj,          // [BS][G] f32 (x-part + bias)
    float* __restrict__ out)                  // outputs [B][S][H], then h_fin, c_fin
{
  __shared__ unsigned short hbuf[2][B_ * H_];   // 2 x 32 KB bf16 hidden state

  const int tid  = threadIdx.x;
  const int lane = tid & 31;
  const int wave = tid >> 5;
  const int half = lane >> 4;
  const int l16  = lane & 15;

  // h0 = 0
  for (int e = tid; e < B_ * H_; e += 1024) hbuf[0][e] = 0;

  // c0 = 0 (persistent per-lane cell state for both groups)
  v8f creg[2];
#pragma unroll
  for (int q = 0; q < 2; ++q)
#pragma unroll
    for (int r = 0; r < 8; ++r) creg[q][r] = 0.0f;

  const size_t OUT_HF = (size_t)B_ * S_ * H_;
  const size_t OUT_CF = OUT_HF + (size_t)B_ * H_;

  // Per-group t-invariant bases (hoisted out of the scan loop).
  const unsigned short* wrow[2][4];
  const float* xpbase[2][4][8];
#pragma unroll
  for (int q = 0; q < 2; ++q) {
    const int gid = wave * 2 + q;
    const int mt  = gid & 3;
    const int js  = gid >> 2;
#pragma unroll
    for (int gg = 0; gg < 4; ++gg) {
      wrow[q][gg] = Whb + (size_t)(gg * H_ + js * 16 + l16) * H_;
#pragma unroll
      for (int r = 0; r < 8; ++r) {
        const int mrow = mt * 16 + r + 8 * half;
        xpbase[q][gg][r] = xproj + (size_t)mrow * S_ * G_ + (gg * H_ + js * 16 + l16);
      }
    }
  }

  __syncthreads();

  for (int t = 0; t < S_; ++t) {
    const unsigned short* hprev = hbuf[t & 1];
    unsigned short* hnext = hbuf[(t + 1) & 1];

#pragma unroll
    for (int q = 0; q < 2; ++q) {
      const int gid = wave * 2 + q;
      const int mt  = gid & 3;        // batch tile: rows mt*16 .. +15
      const int js  = gid >> 2;       // hidden slice: cols js*16 .. +15

      // Launch x_proj loads now; they complete under the K-loop.
      float xp[4][8];
#pragma unroll
      for (int gg = 0; gg < 4; ++gg)
#pragma unroll
        for (int r = 0; r < 8; ++r)
          xp[gg][r] = xpbase[q][gg][r][(size_t)t * G_];

      v8f gacc[4];
#pragma unroll
      for (int gg = 0; gg < 4; ++gg)
#pragma unroll
        for (int r = 0; r < 8; ++r) gacc[gg][r] = 0.0f;

      if (t > 0) {
        const unsigned short* arow = hprev + (mt * 16 + l16) * H_;
#pragma unroll
        for (int kk = 0; kk < H_; kk += 32) {
          v16bf af = frag_a(arow, kk, half);      // h_{t-1} from LDS
          v16bf b0 = frag_b(wrow[q][0], kk, half);
          v16bf b1 = frag_b(wrow[q][1], kk, half);
          v16bf b2 = frag_b(wrow[q][2], kk, half);
          v16bf b3 = frag_b(wrow[q][3], kk, half);
          gacc[0] = __builtin_amdgcn_wmma_f32_16x16x32_bf16(
              false, af, false, b0, (short)0, gacc[0], false, false);
          gacc[1] = __builtin_amdgcn_wmma_f32_16x16x32_bf16(
              false, af, false, b1, (short)0, gacc[1], false, false);
          gacc[2] = __builtin_amdgcn_wmma_f32_16x16x32_bf16(
              false, af, false, b2, (short)0, gacc[2], false, false);
          gacc[3] = __builtin_amdgcn_wmma_f32_16x16x32_bf16(
              false, af, false, b3, (short)0, gacc[3], false, false);
        }
      }

      // LSTM cell update, fully in registers (gate order per reference: f,i,g,o)
#pragma unroll
      for (int r = 0; r < 8; ++r) {
        const int mrow = mt * 16 + r + 8 * half;   // batch index
        const int j    = js * 16 + l16;            // hidden index
        const float f  = sigmoidf_(gacc[0][r] + xp[0][r]);
        const float ii = sigmoidf_(gacc[1][r] + xp[1][r]);
        const float g  = tanhf(gacc[2][r] + xp[2][r]);
        const float o  = sigmoidf_(gacc[3][r] + xp[3][r]);
        const float cn = f * creg[q][r] + ii * g;
        creg[q][r] = cn;
        const float hn = o * tanhf(cn);
        hnext[mrow * H_ + j] = f2bf(hn);
        out[((size_t)mrow * S_ + t) * H_ + j] = hn;
        if (t == S_ - 1) {
          out[OUT_HF + (size_t)mrow * H_ + j] = hn;
          out[OUT_CF + (size_t)mrow * H_ + j] = cn;
        }
      }
    }
    __syncthreads();   // h_t fully in LDS before step t+1 reads it
  }
}

// ---------------- launcher ----------------

extern "C" void kernel_launch(void* const* d_in, const int* in_sizes, int n_in,
                              void* d_out, int out_size, void* d_ws, size_t ws_size,
                              hipStream_t stream) {
  (void)in_sizes; (void)n_in; (void)out_size; (void)ws_size;

  const float* x  = (const float*)d_in[0];   // [B][S][I]
  const float* W  = (const float*)d_in[1];   // [4H][H+I]
  const float* bv = (const float*)d_in[2];   // [4H]
  float* out = (float*)d_out;

  char* ws = (char*)d_ws;
  const size_t XB_BYTES  = (size_t)BS_ * I_ * sizeof(unsigned short);   // 32 MB
  const size_t WHB_BYTES = (size_t)G_ * H_ * sizeof(unsigned short);    // 512 KB
  const size_t WXB_BYTES = (size_t)G_ * I_ * sizeof(unsigned short);    // 512 KB

  unsigned short* xb    = (unsigned short*)(ws);
  unsigned short* whb   = (unsigned short*)(ws + XB_BYTES);
  unsigned short* wxb   = (unsigned short*)(ws + XB_BYTES + WHB_BYTES);
  float*          xproj = (float*)(ws + XB_BYTES + WHB_BYTES + WXB_BYTES); // 256 MB

  const int nx = (int)(BS_ * I_);
  cvt_x_bf16<<<nx / (256 * 4), 256, 0, stream>>>(x, xb, nx);
  cvt_w_bf16<<<(G_ * (H_ + I_)) / 256, 256, 0, stream>>>(W, whb, wxb);

  dim3 g1((unsigned)(BS_ / 64), (unsigned)(G_ / 64));
  gemm_xproj<<<g1, 256, 0, stream>>>(xb, wxb, bv, xproj);

  lstm_scan<<<1, 1024, 0, stream>>>(whb, xproj, out);
}
